// TriangleMultiplicativeUpdate_19138374271356
// MI455X (gfx1250) — matrix operand order repaired and checked by reading, not compiled, not run on hardware
//
#include <hip/hip_runtime.h>

// ---------------- problem constants ----------------
#define NN   512
#define CH   128
#define N2   ((size_t)NN * (size_t)NN)   // 262144
#define EPSF 1e-5f

typedef __bf16 bf16;
typedef __attribute__((ext_vector_type(16))) __bf16 bf16x16;
typedef __attribute__((ext_vector_type(8)))  __bf16 bf16x8;
typedef __attribute__((ext_vector_type(8)))  float  f32x8;

union BF16x16u { bf16x16 v16; bf16x8 v8[2]; };

__device__ __forceinline__ float sigmoidf_(float x) {
  return 1.0f / (1.0f + __expf(-x));
}

// D = A(16x32 bf16) * B(32x16 bf16) + C(16x16 f32)
__device__ __forceinline__ f32x8 wmma_bf16(bf16x16 a, bf16x16 b, f32x8 c) {
  return __builtin_amdgcn_wmma_f32_16x16x32_bf16(
      /*neg_a=*/false, a, /*neg_b=*/false, b,
      /*c_mod=*/(short)0, c, /*reuse_a=*/false, /*reuse_b=*/false);
}

// CDNA5 async copy: 16 bytes per lane, global -> LDS, tracked by ASYNCcnt.
// lds_addr = per-lane LDS byte address (low 32 bits of generic pointer).
__device__ __forceinline__ void async_copy_b128(unsigned lds_addr, const void* gptr) {
  asm volatile("global_load_async_to_lds_b128 %0, %1, off"
               :: "v"(lds_addr), "v"(gptr)
               : "memory");
}
__device__ __forceinline__ void wait_async0() {
  asm volatile("s_wait_asynccnt 0" ::: "memory");
}

// A fragment, 16(M) x 32(K), from row-major storage with row stride ld (elems).
// lane<16: M=lane,    K in {0..7} U {16..23}
// lane>=16: M=lane-16, K in {8..15} U {24..31}
__device__ __forceinline__ bf16x16 load_frag_a(const bf16* base, int ld, int lane) {
  int m  = lane & 15;
  int k0 = (lane & 16) ? 8 : 0;
  const bf16* p = base + m * ld + k0;
  BF16x16u u;
  u.v8[0] = *(const bf16x8*)(p);
  u.v8[1] = *(const bf16x8*)(p + 16);
  return u.v16;
}

// B fragment, 32(K) x 16(N), from "NT" storage: row-major rows = N, cols = K,
// row stride ld. lane<16: N=lane, K in {0..7} U {16..23} ; lane>=16: N=lane-16, K in {8..15} U {24..31}
__device__ __forceinline__ bf16x16 load_frag_b(const bf16* base, int ld, int lane) {
  int n  = lane & 15;
  int k0 = (lane & 16) ? 8 : 0;
  const bf16* p = base + n * ld + k0;
  BF16x16u u;
  u.v8[0] = *(const bf16x8*)(p);
  u.v8[1] = *(const bf16x8*)(p + 16);
  return u.v16;
}

// Per-row layernorm of 8 contiguous f32 per thread; 16 threads per row that
// occupy one 16-lane half of a wave32. Writes bf16 result to LDS row.
__device__ __forceinline__ void ln_row_to_lds(const float* __restrict__ rowp,
                                              const float* __restrict__ w,
                                              const float* __restrict__ b,
                                              bf16* __restrict__ dst, int sub) {
  const float4* p4 = (const float4*)(rowp + sub * 8);
  float4 u0 = p4[0], u1 = p4[1];
  float v[8] = {u0.x, u0.y, u0.z, u0.w, u1.x, u1.y, u1.z, u1.w};
  float s = 0.f, ss = 0.f;
#pragma unroll
  for (int q = 0; q < 8; ++q) { s += v[q]; ss += v[q] * v[q]; }
#pragma unroll
  for (int m = 8; m >= 1; m >>= 1) {
    s  += __shfl_xor(s,  m, 16);
    ss += __shfl_xor(ss, m, 16);
  }
  float mean = s * (1.0f / CH);
  float var  = ss * (1.0f / CH) - mean * mean;
  float rstd = rsqrtf(var + EPSF);
#pragma unroll
  for (int q = 0; q < 8; ++q) {
    int c = sub * 8 + q;
    dst[c] = (bf16)((v[q] - mean) * rstd * w[c] + b[c]);
  }
}

// ---------------------------------------------------------------------------
// Kernel 0: convert 6 weight matrices (f32 [in][out]) -> bf16 transposed [out][in]
// order in wT: 0=w_ap 1=w_ag 2=w_bp 3=w_bg 4=w_g 5=w_z
// ---------------------------------------------------------------------------
__global__ __launch_bounds__(256) void prep_weights_kernel(
    const float* __restrict__ w0, const float* __restrict__ w1,
    const float* __restrict__ w2, const float* __restrict__ w3,
    const float* __restrict__ w4, const float* __restrict__ w5,
    bf16* __restrict__ wT) {
  const float* ws[6] = {w0, w1, w2, w3, w4, w5};
  int m = blockIdx.y;
  int e = blockIdx.x * 256 + threadIdx.x;  // 0..16383
  int i = e >> 7;        // input channel
  int o = e & 127;       // output channel
  wT[m * 16384 + o * 128 + i] = (bf16)ws[m][e];
}

// ---------------------------------------------------------------------------
// Kernel 1: LN(z) + 4 gated projections -> a_cm, b_cm (channel-major bf16)
// one block = 16 rows of z; 8 waves, wave = one 16-col tile x 4 projections
// ---------------------------------------------------------------------------
__global__ __launch_bounds__(256) void proj_kernel(
    const float* __restrict__ z, const bf16* __restrict__ wT,
    const float* __restrict__ b_ap, const float* __restrict__ b_ag,
    const float* __restrict__ b_bp, const float* __restrict__ b_bg,
    const float* __restrict__ lnw, const float* __restrict__ lnb,
    bf16* __restrict__ a_cm, bf16* __restrict__ b_cm) {
  __shared__ bf16 zn[16][136];  // +8 pad -> 272B row stride, staggered banks

  const int tid  = threadIdx.x;
  const int row0 = blockIdx.x * 16;

  {
    int trow = tid >> 4, sub = tid & 15;
    ln_row_to_lds(z + (size_t)(row0 + trow) * CH, lnw, lnb, &zn[trow][0], sub);
  }
  __syncthreads();

  const int wave = tid >> 5, lane = tid & 31;
  const int ct = wave;  // output col tile 0..7

  bf16x16 afrag[4];
#pragma unroll
  for (int kc = 0; kc < 4; ++kc)
    afrag[kc] = load_frag_a(&zn[0][0] + kc * 32, 136, lane);

  f32x8 accs[4];
#pragma unroll
  for (int p = 0; p < 4; ++p) {  // ap, ag, bp, bg
    f32x8 acc = {0.f, 0.f, 0.f, 0.f, 0.f, 0.f, 0.f, 0.f};
    const bf16* wb = wT + p * 16384 + ct * 16 * 128;
#pragma unroll
    for (int kc = 0; kc < 4; ++kc)
      acc = wmma_bf16(afrag[kc], load_frag_b(wb + kc * 32, 128, lane), acc);
    accs[p] = acc;
  }

  const int c  = ct * 16 + (lane & 15);
  const int mo = (lane & 16) ? 8 : 0;
  const float vap = b_ap[c], vag = b_ag[c], vbp = b_bp[c], vbg = b_bg[c];
  bf16x8 av, bv;
#pragma unroll
  for (int v = 0; v < 8; ++v) {
    av[v] = (bf16)((accs[0][v] + vap) * sigmoidf_(accs[1][v] + vag));
    bv[v] = (bf16)((accs[2][v] + vbp) * sigmoidf_(accs[3][v] + vbg));
  }
  *(bf16x8*)(a_cm + (size_t)c * N2 + row0 + mo) = av;
  *(bf16x8*)(b_cm + (size_t)c * N2 + row0 + mo) = bv;
}

// ---------------------------------------------------------------------------
// Kernel 2: per-channel NT GEMM  X_c = A_c * B_c^T   (512x512x512, 128 channels)
// block = one 128x128 output tile of one channel.
// Double-buffered LDS pipeline using CDNA5 async global->LDS copies:
// chunk i+1 streams into buf^1 (ASYNCcnt) while WMMAs consume buf.
// X stored row-major [i][k][c] f32 so the gather kernel reads contiguous rows.
// ---------------------------------------------------------------------------
__global__ __launch_bounds__(256) void einsum_kernel(
    const bf16* __restrict__ a_cm, const bf16* __restrict__ b_cm,
    float* __restrict__ x) {
  __shared__ bf16 As[2][128][40];  // 32 + 8 pad -> 80B stride (16B aligned)
  __shared__ bf16 Bs[2][128][40];

  const int c  = blockIdx.y;
  const int i0 = (blockIdx.x >> 2) * 128;
  const int k0 = (blockIdx.x & 3) * 128;
  const int tid  = threadIdx.x;
  const int wave = tid >> 5, lane = tid & 31;
  const int wr = wave >> 1, wc = wave & 1;  // wave covers 2 row-tiles x 4 col-tiles

  const bf16* Ag = a_cm + (size_t)c * N2;
  const bf16* Bg = b_cm + (size_t)c * N2;

  f32x8 acc[2][4];
#pragma unroll
  for (int r = 0; r < 2; ++r)
#pragma unroll
    for (int q = 0; q < 4; ++q)
      acc[r][q] = (f32x8){0.f, 0.f, 0.f, 0.f, 0.f, 0.f, 0.f, 0.f};

  const int lrow  = tid >> 1;
  const int lhalf = (tid & 1) * 16;

  // issue async fill of one 128x32 A-chunk + B-chunk into buffer `buf`
  auto issue = [&](int buf, int jc) {
    const bf16* ap = Ag + (size_t)(i0 + lrow) * NN + jc + lhalf;
    const bf16* bp = Bg + (size_t)(k0 + lrow) * NN + jc + lhalf;
    unsigned la = (unsigned)(uintptr_t)&As[buf][lrow][lhalf];
    unsigned lb = (unsigned)(uintptr_t)&Bs[buf][lrow][lhalf];
    async_copy_b128(la,      ap);
    async_copy_b128(la + 16, ap + 8);
    async_copy_b128(lb,      bp);
    async_copy_b128(lb + 16, bp + 8);
  };

  issue(0, 0);
  int cur = 0;
  for (int jc = 0; jc < NN; jc += 32) {
    wait_async0();        // my async fills of buf `cur` have landed in LDS
    __syncthreads();      // everyone's fills landed; prev compute reads done

    if (jc + 32 < NN) issue(cur ^ 1, jc + 32);  // overlap next fill w/ compute

    bf16x16 af[2], bfm[4];
#pragma unroll
    for (int r = 0; r < 2; ++r)
      af[r] = load_frag_a(&As[cur][(wr * 2 + r) * 16][0], 40, lane);
#pragma unroll
    for (int q = 0; q < 4; ++q)
      bfm[q] = load_frag_b(&Bs[cur][(wc * 4 + q) * 16][0], 40, lane);
#pragma unroll
    for (int r = 0; r < 2; ++r)
#pragma unroll
      for (int q = 0; q < 4; ++q)
        acc[r][q] = wmma_bf16(af[r], bfm[q], acc[r][q]);

    cur ^= 1;
  }

  const int nn = lane & 15, mo = (lane & 16) ? 8 : 0;
#pragma unroll
  for (int r = 0; r < 2; ++r)
#pragma unroll
    for (int q = 0; q < 4; ++q) {
      const int kk = k0 + (wc * 4 + q) * 16 + nn;
#pragma unroll
      for (int v = 0; v < 8; ++v) {
        const int ii = i0 + (wr * 2 + r) * 16 + mo + v;
        x[((size_t)ii * NN + kk) * CH + c] = acc[r][q][v];
      }
    }
}

// ---------------------------------------------------------------------------
// Kernel 3: gather(dst) + LN(x) @ w_z + b_z, gated by sigmoid(LN(z) @ w_g + b_g)
// ---------------------------------------------------------------------------
__global__ __launch_bounds__(256) void final_kernel(
    const float* __restrict__ z, const float* __restrict__ x,
    const int* __restrict__ dst, const bf16* __restrict__ wT,
    const float* __restrict__ b_g, const float* __restrict__ b_z,
    const float* __restrict__ lniw, const float* __restrict__ lnib,
    const float* __restrict__ lnow, const float* __restrict__ lnob,
    float* __restrict__ out) {
  __shared__ bf16 zn[16][136];
  __shared__ bf16 xn[16][136];

  const int tid  = threadIdx.x;
  const int row0 = blockIdx.x * 16;

  {
    int trow = tid >> 4, sub = tid & 15;
    int r = row0 + trow;
    // gate path input: LN(z) row
    ln_row_to_lds(z + (size_t)r * CH, lniw, lnib, &zn[trow][0], sub);
    // gathered einsum output: x[i, dst[i,j], :]
    int i = r >> 9;  // N = 512
    int kidx = dst[r];
    ln_row_to_lds(x + ((size_t)i * NN + (size_t)kidx) * CH, lnow, lnob,
                  &xn[trow][0], sub);
  }
  __syncthreads();

  const int wave = tid >> 5, lane = tid & 31;
  const int ct = wave;

  f32x8 accz = {0.f, 0.f, 0.f, 0.f, 0.f, 0.f, 0.f, 0.f};
  f32x8 accg = {0.f, 0.f, 0.f, 0.f, 0.f, 0.f, 0.f, 0.f};
  const bf16* wzb = wT + 5 * 16384 + ct * 16 * 128;
  const bf16* wgb = wT + 4 * 16384 + ct * 16 * 128;
#pragma unroll
  for (int kc = 0; kc < 4; ++kc) {
    bf16x16 ax = load_frag_a(&xn[0][0] + kc * 32, 136, lane);
    bf16x16 az = load_frag_a(&zn[0][0] + kc * 32, 136, lane);
    accz = wmma_bf16(ax, load_frag_b(wzb + kc * 32, 128, lane), accz);
    accg = wmma_bf16(az, load_frag_b(wgb + kc * 32, 128, lane), accg);
  }

  const int c  = ct * 16 + (lane & 15);
  const int mo = (lane & 16) ? 8 : 0;
  const float vbz = b_z[c], vbg = b_g[c];
#pragma unroll
  for (int v = 0; v < 8; ++v) {
    float o = (accz[v] + vbz) * sigmoidf_(accg[v] + vbg);
    out[(size_t)(row0 + mo + v) * CH + c] = o;
  }
}

// ---------------------------------------------------------------------------
extern "C" void kernel_launch(void* const* d_in, const int* in_sizes, int n_in,
                              void* d_out, int out_size, void* d_ws, size_t ws_size,
                              hipStream_t stream) {
  const float* z    = (const float*)d_in[0];
  const int*   dst  = (const int*)  d_in[2];   // dst_idx
  const float* w_ap = (const float*)d_in[3];
  const float* b_ap = (const float*)d_in[4];
  const float* w_ag = (const float*)d_in[5];
  const float* b_ag = (const float*)d_in[6];
  const float* w_bp = (const float*)d_in[7];
  const float* b_bp = (const float*)d_in[8];
  const float* w_bg = (const float*)d_in[9];
  const float* b_bg = (const float*)d_in[10];
  const float* w_g  = (const float*)d_in[11];
  const float* b_g  = (const float*)d_in[12];
  const float* w_z  = (const float*)d_in[13];
  const float* b_z  = (const float*)d_in[14];
  const float* lniw = (const float*)d_in[15];
  const float* lnib = (const float*)d_in[16];
  const float* lnow = (const float*)d_in[17];
  const float* lnob = (const float*)d_in[18];
  float* out = (float*)d_out;

  // workspace layout (bytes):
  //   [0)              wT   : 6*128*128 bf16      = 196,608
  //   [196608)         a_cm : 128 * N2 bf16       = 67,108,864
  //   [+64MB)          b_cm : 128 * N2 bf16       = 67,108,864
  //   [+64MB)          x    : N2 * 128 f32        = 134,217,728
  char* ws = (char*)d_ws;
  bf16*  wT   = (bf16*)ws;
  bf16*  a_cm = (bf16*)(ws + 196608);
  bf16*  b_cm = a_cm + (size_t)CH * N2;
  float* xbuf = (float*)(ws + 196608 + 2 * (size_t)67108864);

  prep_weights_kernel<<<dim3(64, 6), 256, 0, stream>>>(
      w_ap, w_ag, w_bp, w_bg, w_g, w_z, wT);

  proj_kernel<<<(int)(N2 / 16), 256, 0, stream>>>(
      z, wT, b_ap, b_ag, b_bp, b_bg, lniw, lnib, a_cm, b_cm);

  einsum_kernel<<<dim3(16, 128), 256, 0, stream>>>(a_cm, b_cm, xbuf);

  final_kernel<<<(int)(N2 / 16), 256, 0, stream>>>(
      z, xbuf, dst, wT, b_g, b_z, lniw, lnib, lnow, lnob, out);
}
